// AttentionAGaLiTeLayer_47150150975893
// MI455X (gfx1250) — matrix operand order repaired and checked by reading, not compiled, not run on hardware
//
#include <hip/hip_runtime.h>
#include <hip/hip_bf16.h>

typedef __bf16 bf16;
typedef __attribute__((ext_vector_type(16))) __bf16 v16bf;
typedef __attribute__((ext_vector_type(8)))  float  v8f;
typedef unsigned int u32x4 __attribute__((ext_vector_type(4)));
typedef int          i32x8 __attribute__((ext_vector_type(8)));
typedef int          i32x4 __attribute__((ext_vector_type(4)));

#define S_LEN 4096
#define D_DIM 1024
#define H_N   8
#define DH    64
#define ETA_N 4
#define R_N   8
#define E_DIM 256        // DH*ETA
#define KQVBG_N 2560     // H*Dh*5
#define P_N   96         // H*ETA*3
#define P_NPAD 128
#define ATTN_N 512       // H*Dh
#define EPS_F 1e-5f
#define PI_F  3.14159265358979323846f

#if __has_builtin(__builtin_amdgcn_tensor_load_to_lds)
#define HAVE_TDM 1
#else
#define HAVE_TDM 0
#endif

// ---------------------------------------------------------------------------
// f32 -> bf16 conversion (grid-stride, row-major preserved)
// ---------------------------------------------------------------------------
__global__ void cvt_f32_bf16_kernel(const float* __restrict__ in,
                                    bf16* __restrict__ out, int n) {
    for (int i = blockIdx.x * blockDim.x + threadIdx.x; i < n;
         i += gridDim.x * blockDim.x) {
        out[i] = (bf16)in[i];
    }
}

// ---------------------------------------------------------------------------
// f32 (K x N) -> bf16 transposed (Npad x K), zero-padded rows [N, Npad).
// 32x32 LDS tile keeps both global sides coalesced. block = 256 (32x8).
// grid = (K/32, Npad/32)
// ---------------------------------------------------------------------------
__global__ __launch_bounds__(256)
void cvt_transpose_bf16_kernel(const float* __restrict__ in,
                               bf16* __restrict__ out,
                               int K, int N, int Npad) {
    __shared__ float tile[32][33];
    const int kb = blockIdx.x * 32;
    const int nb = blockIdx.y * 32;
    const int tx = threadIdx.x & 31;
    const int ty = threadIdx.x >> 5;     // 0..7
    #pragma unroll
    for (int i = 0; i < 32; i += 8) {
        int k = kb + ty + i, n = nb + tx;
        tile[ty + i][tx] = (n < N) ? in[(size_t)k * N + n] : 0.0f;
    }
    __syncthreads();
    #pragma unroll
    for (int i = 0; i < 32; i += 8) {
        int n = nb + ty + i, k = kb + tx;
        out[(size_t)n * K + k] = (bf16)tile[tx][ty + i];
    }
}

// ---------------------------------------------------------------------------
// occil table: occ[t*R + r] = cos((tick0 + t + 1) * omega_r)
// ---------------------------------------------------------------------------
__global__ void occil_kernel(const float* __restrict__ tick,
                             float* __restrict__ occ) {
    int i = blockIdx.x * blockDim.x + threadIdx.x;
    if (i < S_LEN * R_N) {
        int t = i / R_N, r = i % R_N;
        float omega = -PI_F + (2.0f * PI_F / (float)(R_N - 1)) * (float)r;
        occ[i] = cosf((tick[0] + (float)(t + 1)) * omega);
    }
}

// ---------------------------------------------------------------------------
// TDM descriptors per CDNA5 ISA 8.3/8.4.
// Group 1 (dims/strides/tile shape) is loop-invariant; build it once.
// Tile = 128 rows x 32 cols of bf16 (data_size=2B).
// ---------------------------------------------------------------------------
#if HAVE_TDM
__device__ __forceinline__ i32x8 tdm_make_g1_128x32(unsigned cols,
                                                    unsigned rows,
                                                    unsigned stride_elems) {
    i32x8 g1;
    g1[0] = (int)(1u << 16);                       // data_size=1 -> 2 bytes
    g1[1] = (int)((cols & 0xffffu) << 16);         // tensor_dim0[15:0]
    g1[2] = (int)((cols >> 16)                     // tensor_dim0[31:16]
          | ((rows & 0xffffu) << 16));             // tensor_dim1[15:0]
    g1[3] = (int)((rows >> 16)                     // tensor_dim1[31:16]
          | (32u << 16));                          // tile_dim0 = 32
    g1[4] = (int)128u;                             // tile_dim1=128, tile_dim2=0
    g1[5] = (int)stride_elems;                     // tensor_dim0_stride[31:0]
    g1[6] = 0;
    g1[7] = 0;
    return g1;
}

__device__ __forceinline__ void tdm_issue(unsigned lds_off, const bf16* gptr,
                                          const i32x8 g1) {
    unsigned long long ga = (unsigned long long)(uintptr_t)gptr;
    u32x4 g0;
    g0[0] = 1u;                                    // count=1, is_restore=0
    g0[1] = lds_off;                               // lds_addr (bytes)
    g0[2] = (unsigned)(ga & 0xffffffffu);          // global_addr[31:0]
    g0[3] = (unsigned)((ga >> 32) & 0x01ffffffu)   // global_addr[56:32]
          | (2u << 30);                            // type = 2 ("image")
    i32x4 z4 = {0, 0, 0, 0};
#if defined(__clang_major__) && (__clang_major__ >= 23)
    i32x8 z8 = {0, 0, 0, 0, 0, 0, 0, 0};
    __builtin_amdgcn_tensor_load_to_lds(g0, g1, z4, z4, z8, 0);
#else
    __builtin_amdgcn_tensor_load_to_lds(g0, g1, z4, z4, 0);
#endif
}
#endif

// ---------------------------------------------------------------------------
// bf16 WMMA GEMM: C[M,N] = A[M,K](bf16,row) * Bt[Npad,K](bf16,row) + bias[N]
// Bt is B pre-transposed, padded to Npad % 128 == 0.
// Block = 256 threads (8 waves), 128x128 C tile; each wave owns a 32x64
// sub-tile (2x4 WMMA tiles, 8 v_wmma per K-slab). K staged 32-wide in LDS,
// double-buffered and fed by the Tensor Data Mover (overlap DMA/compute).
// ---------------------------------------------------------------------------
__global__ __launch_bounds__(256)
void gemm_bf16_wmma_kernel(const bf16* __restrict__ A,
                           const bf16* __restrict__ Bt,
                           const float* __restrict__ bias,
                           float* __restrict__ C,
                           int M, int N, int K) {
    __shared__ bf16 As[2][128 * 32];   // [buf][m][k]  8KB each
    __shared__ bf16 Bs[2][128 * 32];   // [buf][n][k]

    const int tid  = threadIdx.x;
    const int lane = tid & 31;
    const int wave = tid >> 5;               // 0..7
    const int m0 = blockIdx.y * 128;
    const int n0 = blockIdx.x * 128;

    // wave sub-tile: 32 rows x 64 cols of C
    const int mtb = (wave & 3) * 2;          // m-tile base (2 tiles of 16)
    const int ntb = (wave >> 2) * 4;         // n-tile base (4 tiles of 16)

    v8f acc[2][4];
    #pragma unroll
    for (int i = 0; i < 2; ++i)
        #pragma unroll
        for (int j = 0; j < 4; ++j) acc[i][j] = (v8f){};

    const int col = lane & 15;               // fragment row/col within tile
    const int ka  = (lane >> 4) * 16;        // A frag K chunk base (0 or 16)
    const int kbb = (lane >> 4) * 16;        // B frag K base (0 or 16)

    union Frag { uint4 u[2]; v16bf v; };
    const int nslab = K >> 5;

#if HAVE_TDM
    const unsigned asOff[2] = { (unsigned)(uintptr_t)&As[0][0],
                                (unsigned)(uintptr_t)&As[1][0] };
    const unsigned bsOff[2] = { (unsigned)(uintptr_t)&Bs[0][0],
                                (unsigned)(uintptr_t)&Bs[1][0] };
    const i32x8 g1A = tdm_make_g1_128x32((unsigned)K, (unsigned)(M - m0), (unsigned)K);
    const i32x8 g1B = tdm_make_g1_128x32((unsigned)K, 128u, (unsigned)K);
    const bf16* aTile = A  + (size_t)m0 * K;
    const bf16* bTile = Bt + (size_t)n0 * K;

    if (tid < 32) {                          // wave 0 drives the TDM: slab 0
        tdm_issue(asOff[0], aTile, g1A);
        tdm_issue(bsOff[0], bTile, g1B);
    }
    for (int i = 0; i < nslab; ++i) {
        const int cur = i & 1;
        if (tid < 32) {
            if (i + 1 < nslab) {             // prefetch slab i+1 into other buf
                tdm_issue(asOff[cur ^ 1], aTile + (size_t)(i + 1) * 32, g1A);
                tdm_issue(bsOff[cur ^ 1], bTile + (size_t)(i + 1) * 32, g1B);
                __builtin_amdgcn_s_wait_tensorcnt(2);  // slab i resident
            } else {
                __builtin_amdgcn_s_wait_tensorcnt(0);
            }
        }
        __syncthreads();                     // slab i visible to all waves
        const bf16* as = &As[cur][0];
        const bf16* bs = &Bs[cur][0];
#else
    for (int i = 0; i < nslab; ++i) {
        const int kb = i * 32;
        {
            #pragma unroll
            for (int it = 0; it < 2; ++it) {
                int idx = tid + it * 256;    // 512 uint4 per 128x32 tile
                int row = idx >> 2;
                int cb  = (idx & 3) * 8;
                *reinterpret_cast<uint4*>(&As[0][row * 32 + cb]) =
                    *reinterpret_cast<const uint4*>(A + (size_t)(m0 + row) * K + kb + cb);
                *reinterpret_cast<uint4*>(&Bs[0][row * 32 + cb]) =
                    *reinterpret_cast<const uint4*>(Bt + (size_t)(n0 + row) * K + kb + cb);
            }
        }
        __syncthreads();
        const bf16* as = &As[0][0];
        const bf16* bs = &Bs[0][0];
#endif
        Frag af[2], bf_[4];
        #pragma unroll
        for (int ii = 0; ii < 2; ++ii) {
            af[ii].u[0] = *reinterpret_cast<const uint4*>(
                &as[((mtb + ii) * 16 + col) * 32 + ka]);
            af[ii].u[1] = *reinterpret_cast<const uint4*>(
                &as[((mtb + ii) * 16 + col) * 32 + ka + 8]);
        }
        #pragma unroll
        for (int jj = 0; jj < 4; ++jj) {
            bf_[jj].u[0] = *reinterpret_cast<const uint4*>(
                &bs[((ntb + jj) * 16 + col) * 32 + kbb]);
            bf_[jj].u[1] = *reinterpret_cast<const uint4*>(
                &bs[((ntb + jj) * 16 + col) * 32 + kbb + 8]);
        }
        #pragma unroll
        for (int ii = 0; ii < 2; ++ii)
            #pragma unroll
            for (int jj = 0; jj < 4; ++jj)
                acc[ii][jj] = __builtin_amdgcn_wmma_f32_16x16x32_bf16(
                    false, af[ii].v, false, bf_[jj].v, (short)0,
                    acc[ii][jj], false, false);
        __syncthreads();                     // done reading this buffer
    }

    // ---- epilogue: C/D layout: VGPR j -> row = j + 8*(lane>>4), col = lane&15
    const int rhalf = (lane >> 4) * 8;
    #pragma unroll
    for (int ii = 0; ii < 2; ++ii) {
        #pragma unroll
        for (int jj = 0; jj < 4; ++jj) {
            int gc = n0 + (ntb + jj) * 16 + col;
            if (gc < N) {
                float bv = bias[gc];
                int rbase = m0 + (mtb + ii) * 16 + rhalf;
                #pragma unroll
                for (int j = 0; j < 8; ++j)
                    C[(size_t)(rbase + j) * N + gc] = acc[ii][jj][j] + bv;
            }
        }
    }
}

// ---------------------------------------------------------------------------
// Fused elementwise + sequential scan + einsums.
// grid = H heads, block = 256 threads. Lane `e` owns kc[r][e], sc[e];
// lanes d<64 also own vc[r][d]. 9 dot-products per step via shfl + ds_add_f32.
// ---------------------------------------------------------------------------
__device__ __forceinline__ float sigmoidf_(float x) {
    return 1.0f / (1.0f + __expf(-x));
}

__global__ __launch_bounds__(256)
void agalite_scan_kernel(const float* __restrict__ kqvbg,  // (S, 2560)
                         const float* __restrict__ pbuf,   // (S, 96)
                         const int*   __restrict__ term,   // (S,)
                         const float* __restrict__ occ,    // (S, R)
                         const float* __restrict__ tk0,    // (R,H,E)
                         const float* __restrict__ tv0,    // (R,H,Dh)
                         const float* __restrict__ s0,     // (H,E)
                         bf16* __restrict__ attn)          // (S, 512)
{
    const int h   = blockIdx.x;
    const int tid = threadIdx.x;            // == e in [0,256)
    const int d   = tid >> 2;               // e = d*ETA + eta
    const int eta = tid & 3;

    float kc[R_N], vc[R_N], sc;
    #pragma unroll
    for (int r = 0; r < R_N; ++r)
        kc[r] = tk0[((size_t)r * H_N + h) * E_DIM + tid];
    sc = s0[(size_t)h * E_DIM + tid];
    #pragma unroll
    for (int r = 0; r < R_N; ++r)
        vc[r] = (tid < DH) ? tv0[((size_t)r * H_N + h) * DH + tid] : 0.0f;

    __shared__ float red[16];               // [0..7]=kdq[r], [8]=norm

    for (int t = 0; t < S_LEN; ++t) {
        __syncthreads();                     // prior step's reads of red done
        if (tid < 9) red[tid] = 0.0f;

        const float* kq = kqvbg + (size_t)t * KQVBG_N + h * (5 * DH);
        const float* pp = pbuf  + (size_t)t * P_N     + h * (3 * ETA_N);

        float kd = kq[d];
        float qd = kq[DH + d];
        float gd = kq[4 * DH + d];
        float p1 = pp[eta];
        float p2 = pp[ETA_N + eta];
        float p3 = pp[2 * ETA_N + eta];
        float nt = 1.0f - (float)term[t];

        float key = fmaxf(kd, 0.0f) * fmaxf(p1, 0.0f);
        float qe  = fmaxf(qd, 0.0f) * fmaxf(p2, 0.0f);
        float gam = sigmoidf_(gd) * sigmoidf_(p3);
        float kg  = key * gam;
        float dg  = (1.0f - gam) * nt;

        float oc[R_N];
        #pragma unroll
        for (int r = 0; r < R_N; ++r) oc[r] = occ[(size_t)t * R_N + r];

        float pr[9];
        sc = dg * sc + kg;
        pr[8] = sc * qe;
        #pragma unroll
        for (int r = 0; r < R_N; ++r) {
            kc[r] = dg * kc[r] + kg * oc[r];
            pr[r] = kc[r] * qe;
        }

        // wave32 tree reduction, then per-wave leader accumulates into LDS
        #pragma unroll
        for (int i = 0; i < 9; ++i) {
            float v = pr[i];
            v += __shfl_xor(v, 16, 32);
            v += __shfl_xor(v, 8, 32);
            v += __shfl_xor(v, 4, 32);
            v += __shfl_xor(v, 2, 32);
            v += __shfl_xor(v, 1, 32);
            pr[i] = v;
        }
        __syncthreads();                     // red zeroed
        if ((tid & 31) == 0) {
            #pragma unroll
            for (int i = 0; i < 9; ++i) atomicAdd(&red[i], pr[i]);
        }
        __syncthreads();                     // sums complete

        if (tid < DH) {
            float vd = kq[2 * DH + tid];
            float bd = kq[3 * DH + tid];
            float bs = sigmoidf_(bd);
            float vb = vd * bs;
            float db = (1.0f - bs) * nt;
            float kv = 0.0f;
            #pragma unroll
            for (int r = 0; r < R_N; ++r) {
                vc[r] = db * vc[r] + vb * oc[r];
                kv += vc[r] * red[r];
            }
            float a = kv / (2.0f * (float)R_N * red[8] + EPS_F);
            attn[(size_t)t * ATTN_N + h * DH + tid] = (bf16)a;
        }
    }
}

// ---------------------------------------------------------------------------
// launch
// ---------------------------------------------------------------------------
extern "C" void kernel_launch(void* const* d_in, const int* in_sizes, int n_in,
                              void* d_out, int out_size, void* d_ws, size_t ws_size,
                              hipStream_t stream) {
    (void)in_sizes; (void)n_in; (void)out_size; (void)ws_size;

    const float* inputs = (const float*)d_in[0];
    const int*   term   = (const int*)  d_in[1];
    const float* W_kqv  = (const float*)d_in[2];
    const float* b_kqv  = (const float*)d_in[3];
    const float* W_p    = (const float*)d_in[4];
    const float* b_p    = (const float*)d_in[5];
    const float* W_proj = (const float*)d_in[6];
    const float* b_proj = (const float*)d_in[7];
    const float* tk0    = (const float*)d_in[8];
    const float* tv0    = (const float*)d_in[9];
    const float* s0     = (const float*)d_in[10];
    const float* tick   = (const float*)d_in[11];

    char* ws = (char*)d_ws;
    size_t off = 0;
    auto alloc = [&](size_t bytes) {
        void* p = ws + off;
        off += (bytes + 255) & ~size_t(255);
        return p;
    };

    bf16*  in_bf     = (bf16*) alloc((size_t)S_LEN * D_DIM * 2);
    bf16*  wkqvT_bf  = (bf16*) alloc((size_t)KQVBG_N * D_DIM * 2);   // (2560,1024)
    bf16*  wpT_bf    = (bf16*) alloc((size_t)P_NPAD * D_DIM * 2);    // (128,1024)
    bf16*  wprojT_bf = (bf16*) alloc((size_t)D_DIM * ATTN_N * 2);    // (1024,512)
    float* kqvbg     = (float*)alloc((size_t)S_LEN * KQVBG_N * 4);
    float* pbuf      = (float*)alloc((size_t)S_LEN * P_N * 4);
    bf16*  attn_bf   = (bf16*) alloc((size_t)S_LEN * ATTN_N * 2);
    float* occ       = (float*)alloc((size_t)S_LEN * R_N * 4);

    // precision staging: activations row-major bf16; weights transposed bf16
    cvt_f32_bf16_kernel<<<1024, 256, 0, stream>>>(inputs, in_bf, S_LEN * D_DIM);
    cvt_transpose_bf16_kernel<<<dim3(D_DIM / 32, KQVBG_N / 32), 256, 0, stream>>>(
        W_kqv, wkqvT_bf, D_DIM, KQVBG_N, KQVBG_N);
    cvt_transpose_bf16_kernel<<<dim3(D_DIM / 32, P_NPAD / 32), 256, 0, stream>>>(
        W_p, wpT_bf, D_DIM, P_N, P_NPAD);
    cvt_transpose_bf16_kernel<<<dim3(ATTN_N / 32, D_DIM / 32), 256, 0, stream>>>(
        W_proj, wprojT_bf, ATTN_N, D_DIM, D_DIM);
    occil_kernel<<<(S_LEN * R_N + 255) / 256, 256, 0, stream>>>(tick, occ);

    // GEMM1: kqvbg = inputs @ W_kqv + b_kqv   (4096 x 2560, K=1024)
    gemm_bf16_wmma_kernel<<<dim3(KQVBG_N / 128, S_LEN / 128), 256, 0, stream>>>(
        in_bf, wkqvT_bf, b_kqv, kqvbg, S_LEN, KQVBG_N, D_DIM);
    // GEMM2: p = inputs @ W_p + b_p           (4096 x 96, K=1024; Npad=128)
    gemm_bf16_wmma_kernel<<<dim3(P_NPAD / 128, S_LEN / 128), 256, 0, stream>>>(
        in_bf, wpT_bf, b_p, pbuf, S_LEN, P_N, D_DIM);

    // fused scan + attention combine -> attn (bf16)
    agalite_scan_kernel<<<H_N, 256, 0, stream>>>(
        kqvbg, pbuf, term, occ, tk0, tv0, s0, attn_bf);

    // GEMM3: out = attn @ W_proj + b_proj     (4096 x 1024, K=512)
    gemm_bf16_wmma_kernel<<<dim3(D_DIM / 128, S_LEN / 128), 256, 0, stream>>>(
        attn_bf, wprojT_bf, b_proj, (float*)d_out, S_LEN, D_DIM, ATTN_N);
}
